// ReferenceTopKRouter_16217796509890
// MI455X (gfx1250) — matrix operand order repaired and checked by reading, not compiled, not run on hardware
//
#include <hip/hip_runtime.h>

// ---------------------------------------------------------------------------
// MoE top-8 router for MI455X (gfx1250, wave32).
// logits = x @ W^T + b via bf16x3 split-GEMM on v_wmma_f32_16x16x32_bf16.
// One wave32 owns one 16-token tile and ALL 64 experts (4 C tiles): the
// fp32->bf16 hi/lo split of x is done once per tile, and the 12 WMMAs per
// k-step are issued term-major (interleaved across the 4 independent
// accumulators) so dependent WMMAs are 4 issues apart -> no XDL bubbles.
// Fused top-8 + softmax + dense-scatter epilogue, one token per lane.
// ---------------------------------------------------------------------------

typedef __attribute__((ext_vector_type(16))) __bf16          v16bf;
typedef __attribute__((ext_vector_type(16))) unsigned short  v16us;
typedef __attribute__((ext_vector_type(8)))  unsigned        v8u;
typedef __attribute__((ext_vector_type(8)))  float           v8f;

#define DIM      4096
#define EXPERTS  64
#define TOPK     8
#define NEG_INF  (-3.0e38f)
#define PERM_HI2 0x07060302u   // pack {hi16(s1), hi16(s0)} into one dword

// --- pre-pass: W fp32 -> W_hi / W_lo bf16 (1 MB once; W then lives in L2) --
__global__ void convert_weights_kernel(const float* __restrict__ w,
                                       unsigned short* __restrict__ whi,
                                       unsigned short* __restrict__ wlo) {
    int i = blockIdx.x * 256 + threadIdx.x;     // EXPERTS*DIM threads total
    unsigned u  = __builtin_bit_cast(unsigned, w[i]);
    float    fh = __builtin_bit_cast(float, u & 0xffff0000u);
    float    rs = w[i] - fh;                    // exact residual
    whi[i] = (unsigned short)(u >> 16);
    wlo[i] = (unsigned short)(__builtin_bit_cast(unsigned, rs) >> 16);
}

// --- main fused kernel: 4 waves/block, each wave = one 16-token tile -------
__global__ __launch_bounds__(128) void router_wmma_kernel(
    const float*          __restrict__ x,      // [N, 4096]
    const unsigned short* __restrict__ whi,    // [64, 4096] bf16 bits
    const unsigned short* __restrict__ wlo,    // [64, 4096] bf16 bits
    const float*          __restrict__ bias,   // [64]
    float*                __restrict__ scores, // [N, 64]
    float*                __restrict__ idxout) // [N, 8] (indices as float)
{
    __shared__ float lg[4][16 * 68];            // per-wave 16x64 logits, pad->68

    const int wave   = threadIdx.x >> 5;
    const int lane   = threadIdx.x & 31;
    const int row    = lane & 15;               // token-in-tile / expert-in-group
    const int hiHalf = lane >> 4;               // 0 or 1
    const int token0 = blockIdx.x * 64 + wave * 16;

    const float* xrow = x + (size_t)(token0 + row) * DIM;
    const int aK0 = hiHalf * 16;                // A fragment K offset (contiguous, matches B)
    const int bK0 = hiHalf * 16;                // B fragment K offset (contiguous, matches A)

    const size_t wroff = (size_t)row * DIM;     // expert (g*16+row) row base
    const unsigned short* whirow = whi + wroff;
    const unsigned short* wlorow = wlo + wroff;

    v8f acc[4] = {v8f{}, v8f{}, v8f{}, v8f{}};

    #pragma unroll 2
    for (int kb = 0; kb < DIM; kb += 32) {
        // ---- A tile: fragment-ordered coalesced b128 loads (raw bits) -----
        unsigned au[16];
        *(uint4*)(&au[0])  = *(const uint4*)(xrow + kb + aK0);
        *(uint4*)(&au[4])  = *(const uint4*)(xrow + kb + aK0 + 4);
        *(uint4*)(&au[8])  = *(const uint4*)(xrow + kb + aK0 + 8);
        *(uint4*)(&au[12]) = *(const uint4*)(xrow + kb + aK0 + 12);

        // ---- bf16 hi/lo split: v_fma_mix residual + v_perm pairwise pack --
        v8u ah, al;
        #pragma unroll
        for (int j = 0; j < 8; ++j) {
            unsigned u0 = au[2 * j], u1 = au[2 * j + 1];
            float f0 = __builtin_bit_cast(float, u0);
            float f1 = __builtin_bit_cast(float, u1);
            float r0 = f0 - __builtin_bit_cast(float, u0 & 0xffff0000u);  // exact
            float r1 = f1 - __builtin_bit_cast(float, u1 & 0xffff0000u);  // exact
            ah[j] = __builtin_amdgcn_perm(u1, u0, PERM_HI2);
            al[j] = __builtin_amdgcn_perm(__builtin_bit_cast(unsigned, r1),
                                          __builtin_bit_cast(unsigned, r0), PERM_HI2);
        }
        const v16bf Ah = __builtin_bit_cast(v16bf, ah);
        const v16bf Al = __builtin_bit_cast(v16bf, al);

        // ---- hoist all B fragments (pre-split bf16, 32B/lane each) --------
        v16bf Bh[4], Bl[4];
        #pragma unroll
        for (int g = 0; g < 4; ++g) {
            const size_t wo = (size_t)g * 16 * DIM + kb + bK0;
            Bh[g] = __builtin_bit_cast(v16bf, *(const v16us*)(whirow + wo));
            Bl[g] = __builtin_bit_cast(v16bf, *(const v16us*)(wlorow + wo));
        }

        // ---- 12 WMMAs, term-major: dependent pairs are 4 issues apart -----
        #pragma unroll
        for (int g = 0; g < 4; ++g)
            acc[g] = __builtin_amdgcn_wmma_f32_16x16x32_bf16(false, Ah, false, Bh[g],
                                                             (short)0, acc[g], false, false);
        #pragma unroll
        for (int g = 0; g < 4; ++g)
            acc[g] = __builtin_amdgcn_wmma_f32_16x16x32_bf16(false, Ah, false, Bl[g],
                                                             (short)0, acc[g], false, false);
        #pragma unroll
        for (int g = 0; g < 4; ++g)
            acc[g] = __builtin_amdgcn_wmma_f32_16x16x32_bf16(false, Al, false, Bh[g],
                                                             (short)0, acc[g], false, false);
    }

    // ---- logits (+bias) to LDS: VGPR r -> M = hiHalf*8+r, N = g*16+row ----
    #pragma unroll
    for (int g = 0; g < 4; ++g) {
        const float bcol = bias[g * 16 + row];
        #pragma unroll
        for (int r = 0; r < 8; ++r) {
            lg[wave][(hiHalf * 8 + r) * 68 + g * 16 + row] = acc[g][r] + bcol;
        }
    }
    __syncthreads();

    // ---- fused top-8 + softmax + dense scatter: one token per lane --------
    if (lane < 16) {
        float* v = &lg[wave][lane * 68];

        float tv[TOPK];
        int   ti[TOPK];
        for (int k = 0; k < TOPK; ++k) {
            float best = NEG_INF;
            int   bi   = 0;
            for (int e = 0; e < EXPERTS; ++e) {
                float val = v[e];
                if (val > best) { best = val; bi = e; }   // lowest idx wins ties
            }
            v[bi] = NEG_INF;
            tv[k] = best;
            ti[k] = bi;
        }

        // softmax over the 8 (tv[0] is the max)
        float s = 0.f, p[TOPK];
        for (int k = 0; k < TOPK; ++k) { p[k] = __expf(tv[k] - tv[0]); s += p[k]; }
        const float inv = 1.0f / s;

        // rebuild dense row in LDS, then contiguous vector stores
        for (int e = 0; e < EXPERTS; ++e) v[e] = 0.0f;
        for (int k = 0; k < TOPK; ++k)   v[ti[k]] = p[k] * inv;

        float* srow = scores + (size_t)(token0 + lane) * EXPERTS;
        #pragma unroll
        for (int e = 0; e < EXPERTS; e += 4) {
            float4 o = { v[e], v[e + 1], v[e + 2], v[e + 3] };
            *(float4*)(srow + e) = o;
        }

        float* irow = idxout + (size_t)(token0 + lane) * TOPK;
        #pragma unroll
        for (int k = 0; k < TOPK; ++k) irow[k] = (float)ti[k];
    }
}

extern "C" void kernel_launch(void* const* d_in, const int* in_sizes, int n_in,
                              void* d_out, int out_size, void* d_ws, size_t ws_size,
                              hipStream_t stream) {
    const float* x    = (const float*)d_in[0];   // hidden_states [4,4096,4096]
    const float* w    = (const float*)d_in[1];   // weight [64,4096]
    const float* bias = (const float*)d_in[2];   // bias [64]

    const int n_tokens = in_sizes[0] / DIM;      // 16384

    float* out_scores = (float*)d_out;                               // [N,64]
    float* out_idx    = out_scores + (size_t)n_tokens * EXPERTS;     // [N,8]

    unsigned short* whi = (unsigned short*)d_ws;                     // 512 KB
    unsigned short* wlo = whi + (size_t)EXPERTS * DIM;               // 512 KB

    convert_weights_kernel<<<(EXPERTS * DIM) / 256, 256, 0, stream>>>(w, whi, wlo);

    router_wmma_kernel<<<dim3(n_tokens / 64), dim3(128), 0, stream>>>(
        x, whi, wlo, bias, out_scores, out_idx);
}
